// NaryTreeLSTM_81406810128821
// MI455X (gfx1250) — matrix operand order repaired and checked by reading, not compiled, not run on hardware
//
#include <hip/hip_runtime.h>
#include <hip/hip_bf16.h>

// ---------------------------------------------------------------------------
// Binary TreeLSTM (DEPTH=16, HID=INP=256, K=2) as a per-level fused
// bf16-WMMA GEMM:  [x | h_c0+h_c1] (n x 512)  @  B (512 x 5*256)  -> gates
// Gate order g: 0=i, 1=o, 2=u, 3=f(child0), 4=f(child1)
// Heavy levels: 16-wave workgroups share one hidden-col tile; B slice staged
// in LDS via GLOBAL_LOAD_ASYNC_TO_LDS (ASYNCcnt) -> 16x less L2 B traffic.
// ---------------------------------------------------------------------------

typedef __attribute__((ext_vector_type(16))) __bf16 v16bf;
typedef __attribute__((ext_vector_type(8)))  float  v8f;
typedef __attribute__((ext_vector_type(4)))  int    v4i;

#define HID     256
#define INP     256
#define KDIM    512        // INP + HID
#define NG      5
#define DEPTH_C 16
#define NHT     (HID / 16) // 16 hidden-column tiles
#define NKC     (KDIM / 32)// 16 K-chunks of 32
#define BTILE_BYTES (NG * NKC * 32 * 16 * 2)   // 81920 B = one ht's B slice

#if __has_builtin(__builtin_amdgcn_global_load_async_to_lds_b128)
#define USE_ASYNC_LDS 1
#else
#define USE_ASYNC_LDS 0
#endif

__device__ __forceinline__ float sigf(float x) { return 1.0f / (1.0f + __expf(-x)); }

__device__ __forceinline__ void cp_b128_to_lds(const __bf16* gsrc, __bf16* ldst)
{
#if USE_ASYNC_LDS
    // Prototype (from clang diagnostic): (int4 AS1*, int4 AS3*, imm offset, imm cpol)
    __builtin_amdgcn_global_load_async_to_lds_b128(
        (__attribute__((address_space(1))) v4i*)gsrc,
        (__attribute__((address_space(3))) v4i*)ldst, 0, 0);
#else
    *(uint4*)ldst = *(const uint4*)gsrc;     // global_load_b128 + ds_store_b128
#endif
}

__device__ __forceinline__ void wait_async_lds()
{
#if USE_ASYNC_LDS
#if __has_builtin(__builtin_amdgcn_s_wait_asynccnt)
    __builtin_amdgcn_s_wait_asynccnt(0);
#else
    asm volatile("s_wait_asynccnt 0" ::: "memory");
#endif
#endif
}

// ---------------------------------------------------------------------------
// Pack the 5 combined gate weight matrices into WMMA B-fragment order (bf16).
// B logical element B[k][col]:  k<256 -> Wg[col][k]  else Ug[col][k-256]
// B frag layout (32x16 tile): col = lane&15, element e -> K = 16*(lane>>4)+e.
// Storage: Bpk[((g*NHT+ht)*NKC+kc)*32*16 + lane*16 + e];  bias: biasPk[g*256+h].
// ---------------------------------------------------------------------------
__global__ void pack_B_kernel(const float* __restrict__ Wi, const float* __restrict__ Wf,
                              const float* __restrict__ Wo, const float* __restrict__ Wu,
                              const float* __restrict__ Ui, const float* __restrict__ Uo,
                              const float* __restrict__ Uu, const float* __restrict__ Wfk,
                              const float* __restrict__ bi, const float* __restrict__ bf,
                              const float* __restrict__ bo, const float* __restrict__ bu,
                              __bf16* __restrict__ Bpk, float* __restrict__ biasPk)
{
    int tid = blockIdx.x * blockDim.x + threadIdx.x;
    const int total = NG * NHT * NKC * 32;
    if (tid < total) {
        int lane = tid & 31;
        int kc   = (tid >> 5) & 15;
        int ht   = (tid >> 9) & 15;
        int g    = tid >> 13;
        int col  = ht * 16 + (lane & 15);

        const float* Wp;
        const float* Up;
        if      (g == 0) { Wp = Wi; Up = Ui; }
        else if (g == 1) { Wp = Wo; Up = Uo; }
        else if (g == 2) { Wp = Wu; Up = Uu; }
        else if (g == 3) { Wp = Wf; Up = Wfk; }                 // Wfk[0]
        else             { Wp = Wf; Up = Wfk + HID * HID; }     // Wfk[1]

        __bf16 vals[16];
#pragma unroll
        for (int e = 0; e < 16; ++e) {
            int k = kc * 32 + ((lane >> 4) * 16) + e;
            float v = (k < INP) ? Wp[(size_t)col * INP + k]
                                : Up[(size_t)col * HID + (k - INP)];
            vals[e] = (__bf16)v;
        }
        __bf16* dst = Bpk + ((size_t)((g * NHT + ht) * NKC + kc) * 32 + lane) * 16;
#pragma unroll
        for (int e = 0; e < 16; ++e) dst[e] = vals[e];
    }
    if (tid < NG * HID) {
        int g = tid / HID, h = tid % HID;
        const float* bp = (g == 0) ? bi : (g == 1) ? bo : (g == 2) ? bu : bf;
        biasPk[tid] = bp[h];
    }
}

// ---------------------------------------------------------------------------
// Build A (n x 512) for one level in WMMA A-fragment order (bf16).
// A[row][k] = k<256 ? x[row][k] : h_prev[2*row][k-256] + h_prev[2*row+1][k-256]
// A frag layout (16x32): row = lane&15, elem e -> K = 16*(lane>>4)+e
// ---------------------------------------------------------------------------
__global__ void prep_A_kernel(const float* __restrict__ xs, const __bf16* __restrict__ h_prev,
                              __bf16* __restrict__ Afrag, int n, int nTilesM)
{
    int tid = blockIdx.x * blockDim.x + threadIdx.x;
    int total = nTilesM * NKC * 32;
    if (tid >= total) return;
    int lane = tid & 31;
    int kc   = (tid >> 5) & 15;
    int mt   = tid >> 9;
    int row  = mt * 16 + (lane & 15);
    bool valid = row < n;
    int kb = (lane >> 4) * 16;

    __bf16 vals[16];
#pragma unroll
    for (int e = 0; e < 16; ++e) {
        int k = kc * 32 + kb + e;
        float v = 0.0f;
        if (valid) {
            if (k < INP) {
                v = xs[(size_t)row * INP + k];
            } else if (h_prev) {
                int hh = k - INP;
                v = (float)h_prev[(size_t)(2 * row) * HID + hh] +
                    (float)h_prev[(size_t)(2 * row + 1) * HID + hh];
            }
        }
        vals[e] = (__bf16)v;
    }
    __bf16* dst = Afrag + ((size_t)(mt * NKC + kc) * 32 + lane) * 16;
#pragma unroll
    for (int e = 0; e < 16; ++e) dst[e] = vals[e];
}

// ---------------------------------------------------------------------------
// Heavy-level kernel: 512 threads = 16 waves, all sharing one ht tile.
// B slice for this ht staged in LDS (80KB dynamic shared), A streamed from
// global. Wave i handles node tile mt = mtg*16 + i. No divergence -> EXEC
// all-ones for every WMMA.
// ---------------------------------------------------------------------------
__global__ void __launch_bounds__(512)
lstm_level_lds_kernel(const __bf16* __restrict__ Afrag, const __bf16* __restrict__ Bpk,
                      const float* __restrict__ biasPk, const float* __restrict__ c_prev,
                      __bf16* __restrict__ h_out, float* __restrict__ c_out,
                      int mtGroups)
{
    extern __shared__ __align__(16) __bf16 sB[];   // NG*NKC*32*16 bf16 = 80KB
    int ht  = blockIdx.x / mtGroups;
    int mtg = blockIdx.x % mtGroups;
    int tid = threadIdx.x;

    // Stage this ht's B slice: NG contiguous 16KB chunks -> 2560 x 32B blocks.
    for (int b = tid; b < NG * NKC * 32; b += 512) {
        int g  = b >> 9;          // 512 blocks per gate chunk
        int of = b & 511;
        const __bf16* src = Bpk + (size_t)((g * NHT + ht) * NKC) * 512 + (size_t)of * 16;
        __bf16* dst = sB + ((size_t)g * 512 + of) * 16;
        cp_b128_to_lds(src, dst);
        cp_b128_to_lds(src + 8, dst + 8);
    }
    wait_async_lds();
    __syncthreads();

    int lane = tid & 31;
    int wv   = tid >> 5;
    int mt   = mtg * 16 + wv;

    v8f acc0 = {}, acc1 = {}, acc2 = {}, acc3 = {}, acc4 = {};
    const v16bf* A  = (const v16bf*)Afrag + (size_t)mt * NKC * 32 + lane;
    const v16bf* Bs = (const v16bf*)sB + lane;

#pragma unroll 4
    for (int kc = 0; kc < NKC; ++kc) {
        v16bf a  = A[(size_t)kc * 32];
        v16bf b0 = Bs[(0 * NKC + kc) * 32];
        v16bf b1 = Bs[(1 * NKC + kc) * 32];
        v16bf b2 = Bs[(2 * NKC + kc) * 32];
        v16bf b3 = Bs[(3 * NKC + kc) * 32];
        v16bf b4 = Bs[(4 * NKC + kc) * 32];
        acc0 = __builtin_amdgcn_wmma_f32_16x16x32_bf16(false, a, false, b0, (short)0, acc0, false, false);
        acc1 = __builtin_amdgcn_wmma_f32_16x16x32_bf16(false, a, false, b1, (short)0, acc1, false, false);
        acc2 = __builtin_amdgcn_wmma_f32_16x16x32_bf16(false, a, false, b2, (short)0, acc2, false, false);
        acc3 = __builtin_amdgcn_wmma_f32_16x16x32_bf16(false, a, false, b3, (short)0, acc3, false, false);
        acc4 = __builtin_amdgcn_wmma_f32_16x16x32_bf16(false, a, false, b4, (short)0, acc4, false, false);
    }

    // Epilogue. C/D layout: element r -> row = 8*(lane>>4)+r, col = lane&15.
    int col = ht * 16 + (lane & 15);
    float bI = biasPk[0 * HID + col];
    float bO = biasPk[1 * HID + col];
    float bU = biasPk[2 * HID + col];
    float bF = biasPk[3 * HID + col];
    int mbase = mt * 16 + (lane >> 4) * 8;

#pragma unroll
    for (int r = 0; r < 8; ++r) {
        int m = mbase + r;
        float iv = sigf(acc0[r] + bI);
        float ov = sigf(acc1[r] + bO);
        float uv = tanhf(acc2[r] + bU);
        float f0 = sigf(acc3[r] + bF);
        float f1 = sigf(acc4[r] + bF);
        float cc0 = 0.0f, cc1 = 0.0f;
        if (c_prev) {
            cc0 = c_prev[(size_t)(2 * m) * HID + col];
            cc1 = c_prev[(size_t)(2 * m + 1) * HID + col];
        }
        float c = iv * uv + f0 * cc0 + f1 * cc1;
        float h = ov * tanhf(c);
        c_out[(size_t)m * HID + col] = c;
        h_out[(size_t)m * HID + col] = (__bf16)h;
    }
}

// ---------------------------------------------------------------------------
// Tail-level kernel (n <= 128 nodes per tail level): one wave
// per 16x16 tile, B straight from L2; handles store masking + root output.
// ---------------------------------------------------------------------------
__global__ void __launch_bounds__(128)
lstm_level_kernel(const __bf16* __restrict__ Afrag, const __bf16* __restrict__ Bpk,
                  const float* __restrict__ biasPk, const float* __restrict__ c_prev,
                  __bf16* __restrict__ h_out, float* __restrict__ c_out,
                  float* __restrict__ root_out, int n, int nTilesM)
{
    int lane = threadIdx.x & 31;
    int w = blockIdx.x * (blockDim.x >> 5) + (threadIdx.x >> 5);
    int totalWaves = nTilesM * NHT;
    if (w >= totalWaves) return;          // wave-uniform
    int mt = w >> 4;
    int ht = w & 15;

    v8f acc0 = {}, acc1 = {}, acc2 = {}, acc3 = {}, acc4 = {};
    const v16bf* A = (const v16bf*)Afrag + (size_t)mt * NKC * 32 + lane;
    const v16bf* B = (const v16bf*)Bpk + lane;

#pragma unroll 4
    for (int kc = 0; kc < NKC; ++kc) {
        v16bf a  = A[(size_t)kc * 32];
        v16bf b0 = B[(size_t)((0 * NHT + ht) * NKC + kc) * 32];
        v16bf b1 = B[(size_t)((1 * NHT + ht) * NKC + kc) * 32];
        v16bf b2 = B[(size_t)((2 * NHT + ht) * NKC + kc) * 32];
        v16bf b3 = B[(size_t)((3 * NHT + ht) * NKC + kc) * 32];
        v16bf b4 = B[(size_t)((4 * NHT + ht) * NKC + kc) * 32];
        acc0 = __builtin_amdgcn_wmma_f32_16x16x32_bf16(false, a, false, b0, (short)0, acc0, false, false);
        acc1 = __builtin_amdgcn_wmma_f32_16x16x32_bf16(false, a, false, b1, (short)0, acc1, false, false);
        acc2 = __builtin_amdgcn_wmma_f32_16x16x32_bf16(false, a, false, b2, (short)0, acc2, false, false);
        acc3 = __builtin_amdgcn_wmma_f32_16x16x32_bf16(false, a, false, b3, (short)0, acc3, false, false);
        acc4 = __builtin_amdgcn_wmma_f32_16x16x32_bf16(false, a, false, b4, (short)0, acc4, false, false);
    }

    int col = ht * 16 + (lane & 15);
    float bI = biasPk[0 * HID + col];
    float bO = biasPk[1 * HID + col];
    float bU = biasPk[2 * HID + col];
    float bF = biasPk[3 * HID + col];
    int mbase = mt * 16 + (lane >> 4) * 8;

#pragma unroll
    for (int r = 0; r < 8; ++r) {
        int m = mbase + r;
        float iv = sigf(acc0[r] + bI);
        float ov = sigf(acc1[r] + bO);
        float uv = tanhf(acc2[r] + bU);
        float f0 = sigf(acc3[r] + bF);
        float f1 = sigf(acc4[r] + bF);
        float cc0 = 0.0f, cc1 = 0.0f;
        if (c_prev) {
            cc0 = c_prev[(size_t)(2 * m) * HID + col];
            cc1 = c_prev[(size_t)(2 * m + 1) * HID + col];
        }
        float c = iv * uv + f0 * cc0 + f1 * cc1;
        float h = ov * tanhf(c);
        if (m < n) {
            c_out[(size_t)m * HID + col] = c;
            h_out[(size_t)m * HID + col] = (__bf16)h;
            if (root_out && m == 0) {       // depth 0: emit stack([h, c])
                root_out[col]       = h;
                root_out[HID + col] = c;
            }
        }
    }
}

// ---------------------------------------------------------------------------
// Host-side orchestration: pack weights once, then walk the tree bottom-up.
// ---------------------------------------------------------------------------
extern "C" void kernel_launch(void* const* d_in, const int* in_sizes, int n_in,
                              void* d_out, int out_size, void* d_ws, size_t ws_size,
                              hipStream_t stream)
{
    const float* x   = (const float*)d_in[0];
    const float* Wi  = (const float*)d_in[1];
    const float* bi  = (const float*)d_in[2];
    const float* Wf  = (const float*)d_in[3];
    const float* bf  = (const float*)d_in[4];
    const float* Wo  = (const float*)d_in[5];
    const float* bo  = (const float*)d_in[6];
    const float* Wu  = (const float*)d_in[7];
    const float* bu  = (const float*)d_in[8];
    const float* Ui  = (const float*)d_in[9];
    const float* Uo  = (const float*)d_in[10];
    const float* Uu  = (const float*)d_in[11];
    const float* Wfk = (const float*)d_in[12];
    (void)in_sizes; (void)n_in; (void)out_size;

    const size_t NLEAF = (size_t)1 << DEPTH_C;     // 65536
    char* ws = (char*)d_ws;
    size_t off = 0;
    __bf16* Bpk    = (__bf16*)(ws + off); off += (size_t)NG * KDIM * HID * sizeof(__bf16);
    float*  biasPk = (float*)(ws + off);  off += (size_t)NG * HID * sizeof(float);
    off = (off + 255) & ~(size_t)255;
    __bf16* Afrag  = (__bf16*)(ws + off); off += NLEAF * KDIM * sizeof(__bf16);
    __bf16* h_even = (__bf16*)(ws + off); off += NLEAF * HID * sizeof(__bf16);
    __bf16* h_odd  = (__bf16*)(ws + off); off += (NLEAF / 2) * HID * sizeof(__bf16);
    float*  c_even = (float*)(ws + off);  off += NLEAF * HID * sizeof(float);
    float*  c_odd  = (float*)(ws + off);  off += (NLEAF / 2) * HID * sizeof(float);
    (void)ws_size;

    {
        int threads = NG * NHT * NKC * 32;
        pack_B_kernel<<<(threads + 255) / 256, 256, 0, stream>>>(
            Wi, Wf, Wo, Wu, Ui, Uo, Uu, Wfk, bi, bf, bo, bu, Bpk, biasPk);
    }

    for (int d = DEPTH_C; d >= 0; --d) {
        int n = 1 << d;
        const float* xs = x + (size_t)((1 << d) - 1) * INP;
        int nTilesM = (n + 15) / 16;

        __bf16* hcur = (d & 1) ? h_odd : h_even;
        float*  ccur = (d & 1) ? c_odd : c_even;
        const __bf16* hprev = (d == DEPTH_C) ? nullptr : (((d + 1) & 1) ? h_odd : h_even);
        const float*  cprev = (d == DEPTH_C) ? nullptr : (((d + 1) & 1) ? c_odd : c_even);

        int prepThreads = nTilesM * NKC * 32;
        prep_A_kernel<<<(prepThreads + 255) / 256, 256, 0, stream>>>(xs, hprev, Afrag, n, nTilesM);

        if (nTilesM >= 16) {
            // Heavy levels (d >= 8): LDS-staged B, 16 waves per block sharing ht.
            int mtGroups = nTilesM / 16;
            lstm_level_lds_kernel<<<mtGroups * NHT, 512, BTILE_BYTES, stream>>>(
                Afrag, Bpk, biasPk, cprev, hcur, ccur, mtGroups);
        } else {
            int waves  = nTilesM * NHT;
            int blocks = (waves + 3) / 4;
            lstm_level_kernel<<<blocks, 128, 0, stream>>>(
                Afrag, Bpk, biasPk, cprev, hcur, ccur,
                (d == 0) ? (float*)d_out : nullptr, n, nTilesM);
        }
    }
}
